// YoloLoss_48979807043824
// MI455X (gfx1250) — compile-verified, matrix operand order checked by reading, not run on hardware
//
#include <hip/hip_runtime.h>

// ---------------- problem constants ----------------
constexpr int BATCH   = 16;
constexpr int GRID_G  = 52;
constexpr int ANCH    = 3;
constexpr int DEPTH   = 85;                       // 5 + 80 classes
constexpr int CELLS   = GRID_G * GRID_G * ANCH;   // 8112 cells / batch
constexpr int EPB     = CELLS * DEPTH;            // 689,520 floats / batch
constexpr int V4PB    = EPB / 4;                  // 172,380 float4 / batch
constexpr int THREADS = 256;                      // 8 waves (wave32)
constexpr int BPB     = 64;                       // blocks per batch
constexpr int NBLK    = BATCH * BPB;              // 1024 blocks total
constexpr float NCLS  = 16.0f * 2704.0f * 3.0f * 80.0f; // 10,383,360
constexpr float NOBJ  = 16.0f * 2704.0f * 3.0f;         // 129,792

typedef __attribute__((ext_vector_type(16))) _Float16 v16h;
typedef __attribute__((ext_vector_type(8)))  float    v8f;

// ---------- kernel 1: streaming pass, per-block partials -> ws ----------
// Per element (pred x, true t), with e = exp(-x) computed once:
//   BCE(sigmoid(x), t):  log p = -log(1+e), log(1-p) = log p - x  (clamp -100)
//   regression path reuses e:  sigmoid = 1/(1+e)
__global__ void __launch_bounds__(THREADS)
yolo_stream_k1(const float* __restrict__ pred, const float* __restrict__ ytr,
               const float* __restrict__ anch, float4* __restrict__ part) {
    const int b = blockIdx.x >> 6;          // /BPB
    const int j = blockIdx.x & (BPB - 1);
    const float*  pB = pred + (size_t)b * EPB;
    const float*  tB = ytr  + (size_t)b * EPB;
    const float4* p4 = (const float4*)pB;
    const float4* t4 = (const float4*)tB;

    float reg = 0.0f, sobj = 0.0f, clsb = 0.0f, objb = 0.0f;

    for (int u = j * THREADS + threadIdx.x; u < V4PB; u += BPB * THREADS) {
        int nxt = u + BPB * THREADS;
        if (nxt < V4PB) {                    // gfx1250 global_prefetch_b8
            __builtin_prefetch(&p4[nxt], 0, 3);
            __builtin_prefetch(&t4[nxt], 0, 3);
        }
        float4 xp = p4[u];
        float4 xt = t4[u];

        // channel/cell for the first lane of this vector; then increment+wrap
        int flat = u * 4;
        int cell = flat / DEPTH;             // one magic-div per 4 elements
        int ch   = flat - cell * DEPTH;

        #pragma unroll
        for (int i = 0; i < 4; ++i) {
            float x = (i == 0) ? xp.x : (i == 1) ? xp.y : (i == 2) ? xp.z : xp.w;
            float t = (i == 0) ? xt.x : (i == 1) ? xt.y : (i == 2) ? xt.z : xt.w;

            // --- unconditional BCE block (covers 81/85 of live work) ---
            float e   = __expf(-x);
            float lpu = -__logf(1.0f + e);
            float bce = -(t * fmaxf(lpu, -100.0f) +
                          (1.0f - t) * fmaxf(lpu - x, -100.0f));
            bool iscls = (ch >= 5);
            bool isobj = (ch == 4);
            clsb += iscls ? bce : 0.0f;      // v_cndmask, no divergence
            objb += isobj ? bce : 0.0f;
            sobj += isobj ? t   : 0.0f;

            // --- rare regression path (4/85 of elements) ---
            if (ch < 4) {
                float obj = tB[cell * DEPTH + 4];
                float sp  = 1.0f / (1.0f + e);        // reuse e
                int a3    = cell % ANCH;
                float target;
                if (ch < 2) {                         // t_xy = true*G - c_xy
                    int cidx = cell / ANCH;
                    int jj   = cidx % GRID_G;
                    int ii   = cidx / GRID_G;         // meshgrid 'ij': ch0->i, ch1->j
                    target = t * (float)GRID_G - (float)((ch == 0) ? ii : jj);
                } else {                              // t_wh = log(true/anchor)
                    float an    = anch[a3 * 2 + (ch - 2)];
                    float ratio = t / an;
                    target = (ratio > 0.0f) ? __logf(ratio) : 0.0f;
                }
                float d = target - sp;
                reg += d * d * obj;
            }
            // advance channel within the 85-float record
            ++ch;
            if (ch == DEPTH) { ch = 0; ++cell; }
        }
    }

    // wave32 reduction, then cross-wave via LDS (deterministic order)
    #pragma unroll
    for (int off = 16; off > 0; off >>= 1) {
        reg  += __shfl_down(reg,  off, 32);
        sobj += __shfl_down(sobj, off, 32);
        clsb += __shfl_down(clsb, off, 32);
        objb += __shfl_down(objb, off, 32);
    }
    __shared__ float4 s_part[THREADS / 32];
    const int lane = threadIdx.x & 31, wid = threadIdx.x >> 5;
    if (lane == 0) s_part[wid] = make_float4(reg, sobj, clsb, objb);
    __syncthreads();
    if (threadIdx.x == 0) {
        float4 r = s_part[0];
        #pragma unroll
        for (int w = 1; w < THREADS / 32; ++w) {
            r.x += s_part[w].x; r.y += s_part[w].y;
            r.z += s_part[w].z; r.w += s_part[w].w;
        }
        part[blockIdx.x] = r;
    }
}

// ---------- kernel 2: WMMA-based final reduction -> out[16] ----------
__global__ void __launch_bounds__(256)
yolo_reduce_k2(const float4* __restrict__ part, float* __restrict__ out) {
    __shared__ float s_reg[16][16];
    __shared__ float s_sob[16][16];
    __shared__ float s_c[256];
    __shared__ float s_o[256];
    __shared__ float s_scalar;
    const int tid = threadIdx.x;

    {   // per-batch partials: batch b owns blocks [b*64, b*64+64)
        int b = tid >> 4, slot = tid & 15;
        float r = 0.0f, s = 0.0f;
        #pragma unroll
        for (int k = 0; k < 4; ++k) {
            float4 p = part[b * BPB + slot + 16 * k];
            r += p.x; s += p.y;
        }
        s_reg[b][slot] = r; s_sob[b][slot] = s;
    }
    {   // global BCE sums over all 1024 blocks
        float c = 0.0f, o = 0.0f;
        #pragma unroll
        for (int k = 0; k < 4; ++k) {
            float4 p = part[tid + 256 * k];
            c += p.z; o += p.w;
        }
        s_c[tid] = c; s_o[tid] = o;
    }
    __syncthreads();
    for (int s = 128; s > 0; s >>= 1) {
        if (tid < s) { s_c[tid] += s_c[tid + s]; s_o[tid] += s_o[tid + s]; }
        __syncthreads();
    }
    if (tid == 0)
        s_scalar = 0.5f * (s_c[0] * (1.0f / NCLS)) + s_o[0] * (1.0f / NOBJ);
    __syncthreads();

    // Wave 0 reduces the 16x16 partial matrices with one WMMA each:
    // D = A(16x32 f16) x Ones(32x16 f16) + 0, f32 accumulate.
    // A row m: K0..15 = hi-f16 of s[m][0..15] (scaled 2^-8), K16..31 = lo residual.
    if (tid < 32) {
        const float SC = 1.0f / 256.0f, ISC = 256.0f;
        const int row   = tid & 15;
        const int kbase = (tid < 16) ? 0 : 8;   // lanes 16-31 carry K8..15 / K24..31
        v16h areg, asob, bones;
        #pragma unroll
        for (int k = 0; k < 8; ++k) {
            float vr = s_reg[row][kbase + k] * SC;
            float vs = s_sob[row][kbase + k] * SC;
            _Float16 hr = (_Float16)vr;
            _Float16 hs = (_Float16)vs;
            areg[k]     = hr;  areg[8 + k] = (_Float16)(vr - (float)hr);
            asob[k]     = hs;  asob[8 + k] = (_Float16)(vs - (float)hs);
            bones[k]     = (_Float16)1.0f;
            bones[8 + k] = (_Float16)1.0f;
        }
        v8f dreg = {}; v8f dsob = {};
        dreg = __builtin_amdgcn_wmma_f32_16x16x32_f16(false, areg, false, bones,
                                                      (short)0, dreg, false, false);
        dsob = __builtin_amdgcn_wmma_f32_16x16x32_f16(false, asob, false, bones,
                                                      (short)0, dsob, false, false);
        // C/D layout: lane 0 holds M=0..7 (col 0), lane 16 holds M=8..15
        float sc = s_scalar;
        if (tid == 0) {
            #pragma unroll
            for (int m = 0; m < 8; ++m)
                out[m] = 2.0f * (dreg[m] * ISC) + (dsob[m] * ISC) * sc;
        } else if (tid == 16) {
            #pragma unroll
            for (int m = 0; m < 8; ++m)
                out[8 + m] = 2.0f * (dreg[m] * ISC) + (dsob[m] * ISC) * sc;
        }
    }
}

extern "C" void kernel_launch(void* const* d_in, const int* in_sizes, int n_in,
                              void* d_out, int out_size, void* d_ws, size_t ws_size,
                              hipStream_t stream) {
    const float* pred = (const float*)d_in[0];   // y_pred (16,52,52,3,85) f32
    const float* ytr  = (const float*)d_in[1];   // y_true same shape
    const float* anch = (const float*)d_in[2];   // (3,2) f32
    float* out = (float*)d_out;                  // (16,) f32
    float4* part = (float4*)d_ws;                // 1024 * 16 B = 16 KB scratch

    yolo_stream_k1<<<NBLK, THREADS, 0, stream>>>(pred, ytr, anch, part);
    yolo_reduce_k2<<<1, 256, 0, stream>>>(part, out);
}
